// EMSA_40699110097468
// MI455X (gfx1250) — matrix operand order repaired and checked by reading, not compiled, not run on hardware
//
#include <hip/hip_runtime.h>
#include <hip/hip_bf16.h>

#define B_  4
#define C_  512
#define H_  56
#define W_  56
#define NH  8
#define HD  64
#define NQ  (H_*W_)      /* 3136 */
#define OH  28
#define OW  28
#define OQ  (OH*OW)      /* 784 */
#define OQP 800          /* padded V columns (multiple of 32, >= OQ) */
#define EPSV 1e-5f

typedef __attribute__((ext_vector_type(16))) _Float16 v16h;
typedef __attribute__((ext_vector_type(8)))  _Float16 v8h;
typedef __attribute__((ext_vector_type(8)))  float    v8f;

// ---------------------------------------------------------------------------
// WMMA helpers (gfx1250, wave32). Layouts per CDNA5 ISA §7.12.2.
// ---------------------------------------------------------------------------
__device__ __forceinline__ v8f wmma_f16(v16h a, v16h b, v8f c) {
  return __builtin_amdgcn_wmma_f32_16x16x32_f16(false, a, false, b, (short)0, c,
                                                false, false);
}

// A-matrix 16x32 fragment from row-major [m][k] (k contiguous), ld f16 elems.
__device__ __forceinline__ v16h frag_ld_a(const _Float16* base, int ld) {
  int lane = threadIdx.x & 31;
  int m = lane & 15, hf = lane >> 4;
  const _Float16* p = base + (size_t)m * ld + hf * 8;
  v8h lo = *(const v8h*)(p);
  v8h hi = *(const v8h*)(p + 16);
  v16h f;
#pragma unroll
  for (int i = 0; i < 8; ++i) { f[i] = lo[i]; f[i + 8] = hi[i]; }
  return f;
}

// B-matrix 32x16 fragment: operand stored row-major [n][k] (we compute X@W^T).
__device__ __forceinline__ v16h frag_ld_b(const _Float16* base, int ld) {
  int lane = threadIdx.x & 31;
  int n = lane & 15, hf = lane >> 4;
  const _Float16* p = base + (size_t)n * ld + hf * 8;
  v8h lo = *(const v8h*)(p);
  v8h hi = *(const v8h*)(p + 16);
  v16h f;
#pragma unroll
  for (int i = 0; i < 8; ++i) { f[i] = lo[i]; f[i + 8] = hi[i]; }
  return f;
}

// ---------------------------------------------------------------------------
// 1) f32 -> f16 weight convert
// ---------------------------------------------------------------------------
__global__ void cvt_f16(const float* __restrict__ src, _Float16* __restrict__ dst, int n) {
  int i = blockIdx.x * 256 + threadIdx.x;
  if (i < n) dst[i] = (_Float16)src[i];
}

// ---------------------------------------------------------------------------
// 2) pack x (B,C,H,W) f32 -> xf (B,nq,C) f16 via LDS-tiled transpose
// ---------------------------------------------------------------------------
__global__ __launch_bounds__(256) void pack_x(const float* __restrict__ x,
                                              _Float16* __restrict__ xf) {
  __shared__ float t[32][33];
  int q0 = blockIdx.x * 32, c0 = blockIdx.y * 32, b = blockIdx.z;
  int tx = threadIdx.x & 31, ty = threadIdx.x >> 5;   // 32 x 8
  const float* xb = x + (size_t)b * C_ * NQ;
#pragma unroll
  for (int i = 0; i < 4; ++i)
    t[ty + i * 8][tx] = xb[(size_t)(c0 + ty + i * 8) * NQ + q0 + tx];
  __syncthreads();
  _Float16* xfb = xf + (size_t)b * NQ * C_;
#pragma unroll
  for (int i = 0; i < 4; ++i)
    xfb[(size_t)(q0 + ty + i * 8) * C_ + c0 + tx] = (_Float16)t[tx][ty + i * 8];
}

// ---------------------------------------------------------------------------
// 3) depthwise 3x3 s2 p1 conv + bias + LayerNorm(C) -> xs (B,oq,C) f16
// ---------------------------------------------------------------------------
__global__ __launch_bounds__(256) void sr_ln(const float* __restrict__ x,
                                             const float* __restrict__ srw,
                                             const float* __restrict__ srb,
                                             const float* __restrict__ lng,
                                             const float* __restrict__ lnb,
                                             _Float16* __restrict__ xs) {
  __shared__ float red[256];
  int o = blockIdx.x, b = blockIdx.y;
  int oh = o / OW, ow = o % OW;
  int tid = threadIdx.x;
  float local[2];
#pragma unroll
  for (int j = 0; j < 2; ++j) {
    int c = tid + j * 256;
    const float* xc = x + ((size_t)b * C_ + c) * NQ;
    const float* wc = srw + c * 9;
    float acc = srb[c];
#pragma unroll
    for (int kh = 0; kh < 3; ++kh) {
      int ih = oh * 2 - 1 + kh;
      if ((unsigned)ih < (unsigned)H_) {
#pragma unroll
        for (int kw = 0; kw < 3; ++kw) {
          int iw = ow * 2 - 1 + kw;
          if ((unsigned)iw < (unsigned)W_) acc += xc[ih * W_ + iw] * wc[kh * 3 + kw];
        }
      }
    }
    local[j] = acc;
  }
  red[tid] = local[0] + local[1];
  __syncthreads();
  for (int st = 128; st; st >>= 1) {
    if (tid < st) red[tid] += red[tid + st];
    __syncthreads();
  }
  float mean = red[0] * (1.0f / 512.0f);
  __syncthreads();
  float d0 = local[0] - mean, d1 = local[1] - mean;
  red[tid] = d0 * d0 + d1 * d1;
  __syncthreads();
  for (int st = 128; st; st >>= 1) {
    if (tid < st) red[tid] += red[tid + st];
    __syncthreads();
  }
  float inv = rsqrtf(red[0] * (1.0f / 512.0f) + EPSV);
  _Float16* outp = xs + ((size_t)b * OQ + o) * C_;
#pragma unroll
  for (int j = 0; j < 2; ++j) {
    int c = tid + j * 256;
    outp[c] = (_Float16)((local[j] - mean) * inv * lng[c] + lnb[c]);
  }
}

// ---------------------------------------------------------------------------
// 4) GEMM  out = A(Mx512 f16) @ W^T(512x512 f16)  via WMMA
//    block = 128 thr (4 waves), tile 64x64; each wave 32x32 (4 accs,
//    2 A-frags x 2 B-frags per k-step -> 1.0 fragment loads per WMMA)
//    MODE 0: f16 out row-major (ld=512)
//    MODE 1: V-projection, store transposed as Vt[b][h][d][o] (ld=OQP)
//    MODE 2: proj epilogue: +bias, f32 NCHW store into d_out
// ---------------------------------------------------------------------------
template <int MODE>
__global__ __launch_bounds__(128) void gemm_xwT(const _Float16* __restrict__ A,
                                                const _Float16* __restrict__ Wt,
                                                _Float16* __restrict__ outH,
                                                float* __restrict__ outF,
                                                const float* __restrict__ bias) {
  int tid = threadIdx.x, lane = tid & 31, w = tid >> 5;
  int m0 = blockIdx.x * 64 + (w >> 1) * 32;
  int n0 = blockIdx.y * 64 + (w & 1) * 32;
  const _Float16* Ab = A + (size_t)m0 * C_;
  const _Float16* Wb = Wt + (size_t)n0 * C_;
  v8f acc00 = {}, acc01 = {}, acc10 = {}, acc11 = {};
#pragma unroll 4
  for (int kt = 0; kt < 16; ++kt) {
    v16h a0 = frag_ld_a(Ab + kt * 32, C_);
    v16h a1 = frag_ld_a(Ab + (size_t)16 * C_ + kt * 32, C_);
    v16h b0 = frag_ld_b(Wb + kt * 32, C_);
    v16h b1 = frag_ld_b(Wb + (size_t)16 * C_ + kt * 32, C_);
    acc00 = wmma_f16(a0, b0, acc00);
    acc01 = wmma_f16(a0, b1, acc01);
    acc10 = wmma_f16(a1, b0, acc10);
    acc11 = wmma_f16(a1, b1, acc11);
  }
  int nl = lane & 15, hf = lane >> 4;
  auto store = [&](const v8f& acc, int mb, int nb) {
#pragma unroll
    for (int r = 0; r < 8; ++r) {
      int m = mb + hf * 8 + r;
      int c = nb + nl;
      if (MODE == 0) {
        outH[(size_t)m * C_ + c] = (_Float16)acc[r];
      } else if (MODE == 1) {
        int b = m / OQ, o = m % OQ;
        outH[(((size_t)b * NH + (c >> 6)) * HD + (c & 63)) * OQP + o] = (_Float16)acc[r];
      } else {
        int b = m / NQ, q = m % NQ;
        outF[((size_t)b * C_ + c) * NQ + q] = acc[r] + bias[c];
      }
    }
  };
  store(acc00, m0, n0);
  store(acc01, m0, n0 + 16);
  store(acc10, m0 + 16, n0);
  store(acc11, m0 + 16, n0 + 16);
}

// ---------------------------------------------------------------------------
// 5) fused S' = (head-mixed, scaled) Q @ K^T + tc_b -> softmax(o) -> P (f16)
//    + per-(b,g) sum/sumsq accumulation for InstanceNorm.
//    A-fragments preloaded & pre-scaled ONCE per wave (packed f16 splat mul,
//    constant indices); inner o-tile loop is pure B-load + WMMA.
// ---------------------------------------------------------------------------
__global__ __launch_bounds__(256) void attn_softmax(const _Float16* __restrict__ Qf,
                                                    const _Float16* __restrict__ Kf,
                                                    const float* __restrict__ tcw,
                                                    const float* __restrict__ tcb,
                                                    _Float16* __restrict__ P,
                                                    float* __restrict__ stats) {
  __shared__ float S[16][OQ + 1];
  __shared__ float blk[2];
  int qt = blockIdx.x, g = blockIdx.y, b = blockIdx.z;
  int tid = threadIdx.x, lane = tid & 31, w = tid >> 5;
  int q0 = qt * 16;
  const float scale = 0.125f;  // hd^-0.5
  float cbias = tcb[g];
  if (tid < 2) blk[tid] = 0.0f;

  const _Float16* Abase = Qf + ((size_t)b * NQ + q0) * C_;
  const _Float16* Bbase0 = Kf + (size_t)b * OQ * C_;
  int nl = lane & 15, hf = lane >> 4;

  // preload 16 A-fragments, folding tc_w[g,h]*scale (each K=32 slice is
  // inside one 64-wide head: h = kt>>1) -> v_pk_mul_f16 splat
  v16h afr[16];
#pragma unroll
  for (int kt = 0; kt < 16; ++kt) {
    _Float16 hs = (_Float16)(tcw[g * NH + (kt >> 1)] * scale);
    afr[kt] = frag_ld_a(Abase + kt * 32, C_) * hs;
  }

  for (int ot = w; ot < OQ / 16; ot += 8) {
    v8f acc = {};
    const _Float16* Bbase = Bbase0 + (size_t)ot * 16 * C_;
#pragma unroll
    for (int kt = 0; kt < 16; ++kt) {
      v16h bm = frag_ld_b(Bbase + kt * 32, C_);
      acc = wmma_f16(afr[kt], bm, acc);
    }
#pragma unroll
    for (int r = 0; r < 8; ++r) S[hf * 8 + r][ot * 16 + nl] = acc[r] + cbias;
  }
  __syncthreads();

  float s1 = 0.0f, s2 = 0.0f;
#pragma unroll
  for (int rr = 0; rr < 2; ++rr) {
    int row = w * 2 + rr;
    float mx = -3.0e38f;
    for (int c = lane; c < OQ; c += 32) mx = fmaxf(mx, S[row][c]);
#pragma unroll
    for (int off = 16; off; off >>= 1) mx = fmaxf(mx, __shfl_xor(mx, off, 32));
    float sum = 0.0f;
    for (int c = lane; c < OQ; c += 32) {
      float e = __expf(S[row][c] - mx);
      S[row][c] = e;
      sum += e;
    }
#pragma unroll
    for (int off = 16; off; off >>= 1) sum += __shfl_xor(sum, off, 32);
    float inv = 1.0f / sum;
    size_t base = ((size_t)(b * NH + g) * NQ + q0 + row) * OQ;
    for (int c = lane; c < OQ; c += 32) {
      float p = S[row][c] * inv;
      P[base + c] = (_Float16)p;
      s1 += p;
      s2 += p * p;
    }
  }
#pragma unroll
  for (int off = 16; off; off >>= 1) {
    s1 += __shfl_xor(s1, off, 32);
    s2 += __shfl_xor(s2, off, 32);
  }
  if (lane == 0) { atomicAdd(&blk[0], s1); atomicAdd(&blk[1], s2); }
  __syncthreads();
  if (tid == 0) {
    atomicAdd(&stats[(b * NH + g) * 4 + 0], blk[0]);
    atomicAdd(&stats[(b * NH + g) * 4 + 1], blk[1]);
  }
}

// ---------------------------------------------------------------------------
// 6) finalize InstanceNorm stats: mean and rsqrt(var+eps) per (b,g)
// ---------------------------------------------------------------------------
__global__ void stats_fin(float* __restrict__ stats) {
  int i = threadIdx.x;
  if (i < B_ * NH) {
    const float n = (float)NQ * (float)OQ;
    float m = stats[i * 4 + 0] / n;
    float var = stats[i * 4 + 1] / n - m * m;
    stats[i * 4 + 2] = m;
    stats[i * 4 + 3] = rsqrtf(var + EPSV);
  }
}

// ---------------------------------------------------------------------------
// 7) out_h = ((P - m)*inv * attn_weights) @ V   via LDS-staged A + WMMA
//    block = 128 thr (4 waves): 16 q-rows x 64 (=hd) cols, K over 784 o.
//    attn_weights is streamed read-once -> non-temporal loads.
// ---------------------------------------------------------------------------
__global__ __launch_bounds__(128) void av_gemm(const _Float16* __restrict__ P,
                                               const float* __restrict__ aw,
                                               const _Float16* __restrict__ Vt,
                                               const float* __restrict__ stats,
                                               _Float16* __restrict__ Xo) {
  __shared__ __align__(16) _Float16 At[16][40];
  int qt = blockIdx.x, h = blockIdx.y, b = blockIdx.z;
  int tid = threadIdx.x, lane = tid & 31, w = tid >> 5;
  int q0 = qt * 16;
  int bh = b * NH + h;
  float m = stats[bh * 4 + 2], inv = stats[bh * 4 + 3];
  size_t pbase = ((size_t)bh * NQ + q0) * OQ;
  const _Float16* Vb = Vt + (size_t)bh * HD * OQP + (size_t)(w * 16) * OQP;
  v8f acc = {};
  for (int o0 = 0; o0 < OQ; o0 += 32) {  // 24 full tiles + 16-wide tail (zero-padded)
#pragma unroll
    for (int j = 0; j < 4; ++j) {
      int idx = tid + j * 128;
      int c = idx & 31, r = idx >> 5;
      int o = o0 + c;
      float a = 0.0f;
      if (o < OQ) {
        size_t e = pbase + (size_t)r * OQ + o;
        float p = (float)P[e];
        float wgt = __builtin_nontemporal_load(&aw[e]);
        a = (p - m) * inv * wgt;
      }
      At[r][c] = (_Float16)a;
    }
    __syncthreads();
    v16h af = frag_ld_a(&At[0][0], 40);
    v16h bf = frag_ld_b(Vb + o0, OQP);  // pad cols [784,800) are zeroed
    acc = wmma_f16(af, bf, acc);
    __syncthreads();
  }
  int nl = lane & 15, hf = lane >> 4;
  size_t xb = ((size_t)b * NQ + q0) * C_ + h * HD + w * 16;
#pragma unroll
  for (int r = 0; r < 8; ++r)
    Xo[xb + (size_t)(hf * 8 + r) * C_ + nl] = (_Float16)acc[r];
}

// ---------------------------------------------------------------------------
// host launcher
// ---------------------------------------------------------------------------
extern "C" void kernel_launch(void* const* d_in, const int* in_sizes, int n_in,
                              void* d_out, int out_size, void* d_ws, size_t ws_size,
                              hipStream_t stream) {
  (void)in_sizes; (void)n_in; (void)out_size; (void)ws_size;
  const float* x   = (const float*)d_in[0];
  const float* aw  = (const float*)d_in[1];
  const float* qw  = (const float*)d_in[2];
  const float* kw  = (const float*)d_in[3];
  const float* vw  = (const float*)d_in[4];
  const float* srw = (const float*)d_in[5];
  const float* srb = (const float*)d_in[6];
  const float* lng = (const float*)d_in[7];
  const float* lnb = (const float*)d_in[8];
  const float* tcw = (const float*)d_in[9];
  const float* tcb = (const float*)d_in[10];
  const float* pw  = (const float*)d_in[11];
  const float* pb  = (const float*)d_in[12];
  float* outp = (float*)d_out;

  char* ws = (char*)d_ws;
  size_t off = 0;
  auto alloc = [&](size_t bytes) -> char* {
    char* p = ws + off;
    off += (bytes + 255) & ~(size_t)255;
    return p;
  };
  _Float16* xf16 = (_Float16*)alloc((size_t)B_ * NQ * C_ * 2);   // packed x
  _Float16* xs16 = (_Float16*)alloc((size_t)B_ * OQ * C_ * 2);   // conv+LN
  _Float16* qf16 = (_Float16*)alloc((size_t)B_ * NQ * C_ * 2);   // Q
  _Float16* kf16 = (_Float16*)alloc((size_t)B_ * OQ * C_ * 2);   // K (flat)
  _Float16* vt16 = (_Float16*)alloc((size_t)B_ * NH * HD * OQP * 2); // V^T padded
  _Float16* Pbuf = (_Float16*)alloc((size_t)B_ * NH * NQ * OQ * 2);  // softmax P
  _Float16* xo16 = (_Float16*)alloc((size_t)B_ * NQ * C_ * 2);   // attn@V out
  _Float16* wq16 = (_Float16*)alloc((size_t)C_ * C_ * 2);
  _Float16* wk16 = (_Float16*)alloc((size_t)C_ * C_ * 2);
  _Float16* wv16 = (_Float16*)alloc((size_t)C_ * C_ * 2);
  _Float16* wp16 = (_Float16*)alloc((size_t)C_ * C_ * 2);
  float*    stats = (float*)alloc((size_t)B_ * NH * 4 * sizeof(float));

  // zero stats accumulators + V^T pad region (graph-capture-safe async memset)
  hipMemsetAsync(stats, 0, (size_t)B_ * NH * 4 * sizeof(float), stream);
  hipMemsetAsync(vt16, 0, (size_t)B_ * NH * HD * OQP * 2, stream);

  const int NW = C_ * C_;
  cvt_f16<<<(NW + 255) / 256, 256, 0, stream>>>(qw, wq16, NW);
  cvt_f16<<<(NW + 255) / 256, 256, 0, stream>>>(kw, wk16, NW);
  cvt_f16<<<(NW + 255) / 256, 256, 0, stream>>>(vw, wv16, NW);
  cvt_f16<<<(NW + 255) / 256, 256, 0, stream>>>(pw, wp16, NW);

  pack_x<<<dim3(NQ / 32, C_ / 32, B_), 256, 0, stream>>>(x, xf16);
  sr_ln<<<dim3(OQ, B_), 256, 0, stream>>>(x, srw, srb, lng, lnb, xs16);

  gemm_xwT<0><<<dim3((B_ * NQ) / 64, C_ / 64), 128, 0, stream>>>(xf16, wq16, qf16, nullptr, nullptr);
  gemm_xwT<0><<<dim3((B_ * OQ) / 64, C_ / 64), 128, 0, stream>>>(xs16, wk16, kf16, nullptr, nullptr);
  gemm_xwT<1><<<dim3((B_ * OQ) / 64, C_ / 64), 128, 0, stream>>>(xs16, wv16, vt16, nullptr, nullptr);

  attn_softmax<<<dim3(NQ / 16, NH, B_), 256, 0, stream>>>(qf16, kf16, tcw, tcb, Pbuf, stats);
  stats_fin<<<1, 32, 0, stream>>>(stats);
  av_gemm<<<dim3(NQ / 16, NH, B_), 128, 0, stream>>>(Pbuf, aw, vt16, stats, xo16);

  gemm_xwT<2><<<dim3((B_ * NQ) / 64, C_ / 64), 128, 0, stream>>>(xo16, wp16, nullptr, outp, pb);
}
